// NES_11716670783855
// MI455X (gfx1250) — matrix operand order repaired and checked by reading, not compiled, not run on hardware
//
#include <hip/hip_runtime.h>
#include <math.h>

typedef _Float16 f16;
typedef __attribute__((ext_vector_type(16))) _Float16 v16h;
typedef __attribute__((ext_vector_type(8)))  _Float16 v8h;
typedef __attribute__((ext_vector_type(8)))  float    v8f;

#define MB   8192
#define DIN  16
#define H    128
#define G4   512   // 4*H
#define FC1N 256
#define ACT  10

// ---------------------------------------------------------------------------
// K0: repack W_hh [512,128] f32 -> f16 B-fragments for v_wmma_f32_16x16x32_f16.
// B tile is 32(K) x 16(N). Order used by this code (contiguous): lanes 0-15 hold K=0..15 of the
// chunk, lanes 16-31 hold K=16..31, 16 f16 elements per lane. The 16 elements
// of each (n, kc, half) are contiguous (32B) so each lane does two b128 loads.
// flat index: (((n*4 + kc)*2 + half)*16 + e), value = W_hh[n*128 + 32*kc+16*half+e]
// ---------------------------------------------------------------------------
__global__ void k_pack_whh(const float* __restrict__ W_hh, f16* __restrict__ Bp) {
  int id = blockIdx.x * blockDim.x + threadIdx.x;
  if (id >= G4 * H) return;
  int n  = id >> 7;
  int k  = id & 127;
  int kc = k >> 5, rem = k & 31, hf = (rem >> 4) & 1, e = rem & 15;
  Bp[(size_t)(((n << 2) + kc) * 2 + hf) * 16 + e] = (f16)W_hh[n * H + k];
}

__global__ void k_init_hmax(float* __restrict__ hm, int n) {
  for (int i = blockIdx.x * blockDim.x + threadIdx.x; i < n;
       i += gridDim.x * blockDim.x)
    hm[i] = -INFINITY;
}

// ---------------------------------------------------------------------------
// K1: per-object input projections, stored f16 in a batch-tile-interleaved
// layout so the LSTM kernel's accumulator init is one b128 load per tile:
//   xz[ ((bt*6 + d)*512 + n)*16 + (b & 15) ]   with bt = b >> 4
// A lane later reads elements (8*sel .. 8*sel+7) = its 8 C-layout rows.
// ---------------------------------------------------------------------------
__global__ void k_xz(const float* __restrict__ x, const float* __restrict__ W_ih,
                     const float* __restrict__ b_ih, const float* __restrict__ b_hh,
                     f16* __restrict__ xz) {
  __shared__ float xs[DIN];
  const int row = blockIdx.x;        // b*6 + d
  const int b   = row / 6;
  const int d   = row - b * 6;
  const int bt  = b >> 4;
  const int bl  = b & 15;
  const int tid = threadIdx.x;       // 128
  if (tid < DIN) xs[tid] = x[row * DIN + tid];
  __syncthreads();
#pragma unroll
  for (int q = 0; q < 4; ++q) {
    const int n = q * 128 + tid;
    float acc = b_ih[n] + b_hh[n];
    const float* wr = W_ih + n * DIN;
#pragma unroll
    for (int j = 0; j < DIN; ++j) acc = fmaf(xs[j], wr[j], acc);
    xz[((size_t)(bt * 6 + d) * G4 + n) * 16 + bl] = (f16)acc;
  }
}

// ---------------------------------------------------------------------------
// K2: LSTM over one subset (blockIdx.y+1 = bitmask) for one 16-row batch tile.
// 4 waves; wave w owns h-columns [32w,32w+32): its 8 accumulator tiles are the
// i/f/g/o gate columns for those h-columns, so the gate math + cell state stay
// in registers in WMMA C-layout. h is exchanged through LDS as f16 in a layout
// that A-fragment loads read with two b128 ds loads per K-chunk.
// ---------------------------------------------------------------------------
__global__ __launch_bounds__(128)
void k_lstm_subsets(const f16* __restrict__ xz, const f16* __restrict__ Bp,
                    float* __restrict__ hmax) {
  __shared__ __align__(16) f16 hsh[16][136];   // 128 + 8 pad halves per row

  const int lane = threadIdx.x & 31;
  const int w    = threadIdx.x >> 5;       // wave 0..3
  const int sel  = (lane >> 4) & 1;        // lane half (K/M split)
  const int ln   = lane & 15;
  const int bt   = blockIdx.x;             // batch tile (16 rows)
  const int r0   = bt << 4;
  const int mask = blockIdx.y + 1;         // 1..63 subset bitmask

  int objs[6];
  int m = 0;
#pragma unroll
  for (int d = 0; d < 6; ++d)
    if ((mask >> d) & 1) objs[m++] = d;

  __builtin_prefetch(Bp + (size_t)(w << 8), 0, 0);   // global_prefetch_b8

  float c[2][8];
#pragma unroll
  for (int t = 0; t < 2; ++t)
#pragma unroll
    for (int r = 0; r < 8; ++r) c[t][r] = 0.0f;

  v8f acc[8];

  for (int step = 0; step < m; ++step) {
    const int obj = objs[step];

    // accumulator init = precomputed input projection.
    // One 16B load per tile: 8 contiguous f16 = this lane's 8 C-layout rows.
#pragma unroll
    for (int ti = 0; ti < 8; ++ti) {
      const int n = (w << 5) + ((ti & 1) << 4) + ((ti >> 1) << 7) + ln;
      const v8h xv = *(const v8h*)(xz + ((size_t)(bt * 6 + obj) * G4 + n) * 16
                                      + (sel << 3));
#pragma unroll
      for (int r = 0; r < 8; ++r) acc[ti][r] = (float)xv[r];
    }

    if (step > 0) {
      __syncthreads();                       // h from previous step ready
#pragma unroll
      for (int kc = 0; kc < 4; ++kc) {
        const int cb = (kc << 5) + (sel << 4);
        const v8h alo = *(const v8h*)&hsh[ln][cb];       // K=cb..cb+7
        const v8h ahi = *(const v8h*)&hsh[ln][cb + 8];   // K=cb+8..cb+15
        const v16h a = __builtin_shufflevector(alo, ahi,
            0, 1, 2, 3, 4, 5, 6, 7, 8, 9, 10, 11, 12, 13, 14, 15);
#pragma unroll
        for (int ti = 0; ti < 8; ++ti) {
          const int n = (w << 5) + ((ti & 1) << 4) + ((ti >> 1) << 7) + ln;
          const v16h bfrag =
              *(const v16h*)(Bp + (size_t)(((n << 2) + kc) * 2 + sel) * 16);
          acc[ti] = __builtin_amdgcn_wmma_f32_16x16x32_f16(
              false, a, false, bfrag, (short)0, acc[ti], false, false);
        }
      }
      __syncthreads();                       // reads done before h rewrite
    }

    // gates: i,f,g,o for this wave's h-columns are acc[0/2/4/6 + t]
    const bool last = (step == m - 1);
#pragma unroll
    for (int t = 0; t < 2; ++t) {
#pragma unroll
      for (int r = 0; r < 8; ++r) {
        const float iv = acc[0 + t][r];
        const float fv = acc[2 + t][r];
        const float gv = acc[4 + t][r];
        const float ov = acc[6 + t][r];
        const float ig = 1.0f / (1.0f + __expf(-iv));
        const float fg = 1.0f / (1.0f + __expf(-fv));
        const float og = 1.0f / (1.0f + __expf(-ov));
        const float cn = fg * c[t][r] + ig * tanhf(gv);
        c[t][r] = cn;
        const float hv = og * tanhf(cn);
        const int col = (w << 5) + (t << 4) + ln;
        const int row = r + (sel << 3);
        if (!last) {
          hsh[row][col] = (f16)hv;           // feeds next step's A fragments
        } else {
          atomicMax(&hmax[(size_t)(r0 + row) * H + col], hv);  // subset max-pool
        }
      }
    }
  }
}

// ---------------------------------------------------------------------------
// K3: fused head: relu(hmax @ fc1_W^T + b) @ fc2_W^T + b -> log_softmax.
// One block (256 threads) per batch row; trivial FLOPs.
// ---------------------------------------------------------------------------
__global__ void k_head(const float* __restrict__ hmax,
                       const float* __restrict__ fc1_W, const float* __restrict__ fc1_b,
                       const float* __restrict__ fc2_W, const float* __restrict__ fc2_b,
                       float* __restrict__ out) {
  __shared__ float hrow[H];
  __shared__ float a1[FC1N];
  __shared__ float zz[ACT];
  const int b = blockIdx.x;
  const int tid = threadIdx.x;   // 256
  if (tid < H) hrow[tid] = hmax[(size_t)b * H + tid];
  __syncthreads();
  {
    float acc = fc1_b[tid];
    const float* wr = fc1_W + (size_t)tid * H;
#pragma unroll 4
    for (int j = 0; j < H; ++j) acc = fmaf(hrow[j], wr[j], acc);
    a1[tid] = fmaxf(acc, 0.0f);
  }
  __syncthreads();
  if (tid < ACT) {
    float acc = fc2_b[tid];
    const float* wr = fc2_W + (size_t)tid * FC1N;
#pragma unroll 4
    for (int j = 0; j < FC1N; ++j) acc = fmaf(a1[j], wr[j], acc);
    zz[tid] = acc;
  }
  __syncthreads();
  if (tid < ACT) {
    float mx = zz[0];
#pragma unroll
    for (int a = 1; a < ACT; ++a) mx = fmaxf(mx, zz[a]);
    float s = 0.0f;
#pragma unroll
    for (int a = 0; a < ACT; ++a) s += __expf(zz[a] - mx);
    out[(size_t)b * ACT + tid] = zz[tid] - mx - logf(s);
  }
}

// ---------------------------------------------------------------------------
extern "C" void kernel_launch(void* const* d_in, const int* in_sizes, int n_in,
                              void* d_out, int out_size, void* d_ws, size_t ws_size,
                              hipStream_t stream) {
  const float* x     = (const float*)d_in[0];
  const float* W_ih  = (const float*)d_in[1];
  const float* W_hh  = (const float*)d_in[2];
  const float* b_ih  = (const float*)d_in[3];
  const float* b_hh  = (const float*)d_in[4];
  const float* fc1_W = (const float*)d_in[5];
  const float* fc1_b = (const float*)d_in[6];
  const float* fc2_W = (const float*)d_in[7];
  const float* fc2_b = (const float*)d_in[8];
  float* out = (float*)d_out;

  // workspace carve: xz (f16, 48MB) | packed W_hh (f16, 128KB) | hmax (f32, 4MB)
  char* ws = (char*)d_ws;
  f16* xz = (f16*)ws;
  size_t off = (size_t)MB * 6 * G4 * sizeof(f16);
  f16* Bp = (f16*)(ws + off);
  off += (size_t)G4 * H * sizeof(f16);
  float* hmax = (float*)(ws + off);

  hipLaunchKernelGGL(k_pack_whh, dim3((G4 * H + 255) / 256), dim3(256), 0, stream,
                     W_hh, Bp);
  hipLaunchKernelGGL(k_init_hmax, dim3(512), dim3(256), 0, stream, hmax, MB * H);
  hipLaunchKernelGGL(k_xz, dim3(MB * 6), dim3(128), 0, stream,
                     x, W_ih, b_ih, b_hh, xz);
  hipLaunchKernelGGL(k_lstm_subsets, dim3(MB / 16, 63), dim3(128), 0, stream,
                     xz, Bp, hmax);
  hipLaunchKernelGGL(k_head, dim3(MB), dim3(FC1N), 0, stream,
                     hmax, fc1_W, fc1_b, fc2_W, fc2_b, out);
}
